// Net_6906307412335
// MI455X (gfx1250) — matrix operand order repaired and checked by reading, not compiled, not run on hardware
//
#include <hip/hip_runtime.h>

// ---------------------------------------------------------------------------
// Types for CDNA5 WMMA (wave32)
// ---------------------------------------------------------------------------
typedef __attribute__((ext_vector_type(16))) __bf16 v16bf;
typedef __attribute__((ext_vector_type(8)))  float  v8f;

__device__ __forceinline__ float waveRedSum(float v) {
    #pragma unroll
    for (int o = 16; o > 0; o >>= 1) v += __shfl_xor(v, o, 32);
    return v;
}
__device__ __forceinline__ float waveRedMax(float v) {
    #pragma unroll
    for (int o = 16; o > 0; o >>= 1) v = fmaxf(v, __shfl_xor(v, o, 32));
    return v;
}

#define NEG_SLOPE 0.2f

// ---------------------------------------------------------------------------
// Fragment loaders.
//  A 16x32 bf16 layout: lane 0-15 -> M=lane (K-half 0), lane 16-31 -> K-half 1;
//    vgpr i<4 -> K = half*8 + 2i, vgpr i>=4 -> K = 16 + half*8 + 2(i-4).
//  B 32x16 bf16 layout: lane&15 = column, same K mapping as A:
//    vgpr i<4 -> K = half*8 + 2i, vgpr i>=4 -> K = 16 + half*8 + 2(i-4).
// G = guarded (K tail); otherwise straight-line unconditional loads.
// ---------------------------------------------------------------------------
template <bool G>
__device__ __forceinline__ v16bf load_fragA(const float* __restrict__ A,
                                            size_t rowOff, int K, int k0, int half)
{
    v16bf a = {};
    #pragma unroll
    for (int i = 0; i < 8; ++i) {
        int kb = (i < 4) ? (half * 8 + 2 * i) : (16 + half * 8 + 2 * (i - 4));
        int k  = k0 + kb;
        float f0, f1;
        if (!G) {
            float2 p = *(const float2*)(A + rowOff + k);   // 8B-aligned (k even)
            f0 = p.x; f1 = p.y;
        } else {
            f0 = (k     < K) ? A[rowOff + k]     : 0.0f;
            f1 = (k + 1 < K) ? A[rowOff + k + 1] : 0.0f;
        }
        a[2 * i]     = (__bf16)f0;
        a[2 * i + 1] = (__bf16)f1;
    }
    return a;
}

template <bool G>
__device__ __forceinline__ v16bf load_fragB(const float* __restrict__ B,
                                            int K, int Nc, int k0, int col, int half)
{
    v16bf b = {};
    #pragma unroll
    for (int i = 0; i < 8; ++i) {
        int k = k0 + ((i < 4) ? (half * 8 + 2 * i) : (16 + half * 8 + 2 * (i - 4)));
        float g0, g1;
        if (!G) {
            g0 = B[(size_t)k * Nc + col];
            g1 = B[(size_t)(k + 1) * Nc + col];
        } else {
            g0 = (k     < K) ? B[(size_t)k * Nc + col]       : 0.0f;
            g1 = (k + 1 < K) ? B[(size_t)(k + 1) * Nc + col] : 0.0f;
        }
        b[2 * i]     = (__bf16)g0;
        b[2 * i + 1] = (__bf16)g1;
    }
    return b;
}

// ---------------------------------------------------------------------------
// Dense GEMM: C[M,Nc] = A[M,K] * B[K,Nc], fp32 in/out, bf16 WMMA inside.
// Each wave owns a 32x32 output macro-tile (2x2 WMMA tiles): A fragments are
// reused across 2 N-tiles and B fragments across 2 M-tiles -> 4 back-to-back
// v_wmma per K step. Main K loop is branch-free; K%32 handled in one guarded
// tail iteration. Requires Nc % 32 == 0 (128 / 256 here).
// ---------------------------------------------------------------------------
__global__ __launch_bounds__(256) void gemm_wmma_bf16(
    const float* __restrict__ A, const float* __restrict__ B,
    float* __restrict__ C, int M, int K, int Nc)
{
    const int lane = threadIdx.x & 31;
    const int wave = threadIdx.x >> 5;
    const int tiles_n = Nc >> 5;                 // 32-wide macro tiles
    const int tiles_m = (M + 31) >> 5;
    const int tile = blockIdx.x * 8 + wave;
    if (tile >= tiles_m * tiles_n) return;       // wave-uniform

    const int tm = tile / tiles_n;
    const int tn = tile - tm * tiles_n;
    const int m0 = tm << 5;
    const int n0 = tn << 5;

    const int half = lane >> 4;                  // 0 or 1
    const int mr   = lane & 15;
    const int nc   = lane & 15;

    // clamp rows instead of predicating loads (M=100000 is a multiple of 32,
    // so the clamp is a no-op in practice but keeps the code safe + branch-free)
    const int    r0   = min(m0 + mr,      M - 1);
    const int    r1   = min(m0 + 16 + mr, M - 1);
    const size_t off0 = (size_t)r0 * K;
    const size_t off1 = (size_t)r1 * K;
    const int    c0   = n0 + nc;
    const int    c1   = n0 + 16 + nc;

    v8f acc00 = {}, acc01 = {}, acc10 = {}, acc11 = {};

    const int KT = K & ~31;                      // full 32-wide K tiles
    for (int k0 = 0; k0 < KT; k0 += 32) {
        // prefetch next A K-tile while this one computes (global_prefetch_b8)
        if (k0 + 32 < KT) {
            __builtin_prefetch(A + off0 + k0 + 32, 0, 3);
            __builtin_prefetch(A + off1 + k0 + 32, 0, 3);
        }
        v16bf a0 = load_fragA<false>(A, off0, K, k0, half);
        v16bf a1 = load_fragA<false>(A, off1, K, k0, half);
        v16bf b0 = load_fragB<false>(B, K, Nc, k0, c0, half);
        v16bf b1 = load_fragB<false>(B, K, Nc, k0, c1, half);
        acc00 = __builtin_amdgcn_wmma_f32_16x16x32_bf16(false, a0, false, b0, (short)0, acc00, false, false);
        acc01 = __builtin_amdgcn_wmma_f32_16x16x32_bf16(false, a0, false, b1, (short)0, acc01, false, false);
        acc10 = __builtin_amdgcn_wmma_f32_16x16x32_bf16(false, a1, false, b0, (short)0, acc10, false, false);
        acc11 = __builtin_amdgcn_wmma_f32_16x16x32_bf16(false, a1, false, b1, (short)0, acc11, false, false);
    }
    if (KT < K) {                                // single guarded tail step
        v16bf a0 = load_fragA<true>(A, off0, K, KT, half);
        v16bf a1 = load_fragA<true>(A, off1, K, KT, half);
        v16bf b0 = load_fragB<true>(B, K, Nc, KT, c0, half);
        v16bf b1 = load_fragB<true>(B, K, Nc, KT, c1, half);
        acc00 = __builtin_amdgcn_wmma_f32_16x16x32_bf16(false, a0, false, b0, (short)0, acc00, false, false);
        acc01 = __builtin_amdgcn_wmma_f32_16x16x32_bf16(false, a0, false, b1, (short)0, acc01, false, false);
        acc10 = __builtin_amdgcn_wmma_f32_16x16x32_bf16(false, a1, false, b0, (short)0, acc10, false, false);
        acc11 = __builtin_amdgcn_wmma_f32_16x16x32_bf16(false, a1, false, b1, (short)0, acc11, false, false);
    }

    // C/D layout: VGPR r, lanes 0-15 -> M=r, lanes 16-31 -> M=r+8
    #pragma unroll
    for (int r = 0; r < 8; ++r) {
        int ra = m0 + r + 8 * half;              // M-tile 0
        int rb = ra + 16;                        // M-tile 1
        if (ra < M) {
            C[(size_t)ra * Nc + c0] = acc00[r];
            C[(size_t)ra * Nc + c1] = acc01[r];
        }
        if (rb < M) {
            C[(size_t)rb * Nc + c0] = acc10[r];
            C[(size_t)rb * Nc + c1] = acc11[r];
        }
    }
}

// ---------------------------------------------------------------------------
// alpha_src[n] = h[n,:] . a_s ; alpha_dst[n] = h[n,:] . a_d  (one wave / node)
// ---------------------------------------------------------------------------
__global__ __launch_bounds__(256) void alpha_proj(
    const float* __restrict__ h, const float* __restrict__ a_s,
    const float* __restrict__ a_d, float* __restrict__ asrc,
    float* __restrict__ adst, int Nn, int D)
{
    const int lane = threadIdx.x & 31;
    const int node = (blockIdx.x * blockDim.x + threadIdx.x) >> 5;
    if (node >= Nn) return;
    float ps = 0.f, pd = 0.f;
    for (int f = lane; f < D; f += 32) {
        float v = h[(size_t)node * D + f];
        ps += v * a_s[f];
        pd += v * a_d[f];
    }
    ps = waveRedSum(ps);
    pd = waveRedSum(pd);
    if (lane == 0) { asrc[node] = ps; adst[node] = pd; }
}

// ---------------------------------------------------------------------------
// CSR-by-destination construction (built once, reused by all 3 layers)
// ---------------------------------------------------------------------------
__global__ void zero_int(int* p, int n) {
    int i = blockIdx.x * blockDim.x + threadIdx.x;
    if (i < n) p[i] = 0;
}
__global__ void deg_count(const long long* __restrict__ dst, int* __restrict__ deg,
                          int E, int Nn)
{
    int e = blockIdx.x * blockDim.x + threadIdx.x;
    if (e >= E + Nn) return;
    int d = (e < E) ? (int)dst[e] : (e - E);   // self-loops appended
    atomicAdd(&deg[d], 1);
}
__global__ __launch_bounds__(1024) void exscan(const int* __restrict__ deg,
                                               int* __restrict__ rowp, int Nn)
{
    __shared__ int sums[1024];
    const int tid = threadIdx.x;
    const int chunk = (Nn + 1023) / 1024;
    const int s0 = tid * chunk;
    const int s1 = min(s0 + chunk, Nn);
    int local = 0;
    for (int i = s0; i < s1; ++i) local += deg[i];
    sums[tid] = local;
    __syncthreads();
    for (int off = 1; off < 1024; off <<= 1) {
        int v = (tid >= off) ? sums[tid - off] : 0;
        __syncthreads();
        sums[tid] += v;
        __syncthreads();
    }
    int run = sums[tid] - local;               // exclusive prefix
    for (int i = s0; i < s1; ++i) { rowp[i] = run; run += deg[i]; }
    if (tid == 1023) rowp[Nn] = sums[1023];
}
__global__ void copy_int(const int* __restrict__ a, int* __restrict__ b, int n) {
    int i = blockIdx.x * blockDim.x + threadIdx.x;
    if (i < n) b[i] = a[i];
}
__global__ void fill_csr(const long long* __restrict__ dst, int* __restrict__ cursor,
                         int* __restrict__ eid, int E, int Nn)
{
    int e = blockIdx.x * blockDim.x + threadIdx.x;
    if (e >= E + Nn) return;
    int d = (e < E) ? (int)dst[e] : (e - E);
    int pos = atomicAdd(&cursor[d], 1);
    eid[pos] = e;
}

// ---------------------------------------------------------------------------
// Segment softmax + weighted aggregation per node. One block of D threads per
// node; thread = feature lane. h gathers are coalesced over features and hit
// L2 (h fits in the 192 MB L2). No global atomics.
// ---------------------------------------------------------------------------
template <int D, bool RELU>
__global__ __launch_bounds__(D) void gat_aggregate(
    const float* __restrict__ h, const float* __restrict__ asrc,
    const float* __restrict__ adst, const long long* __restrict__ srcArr,
    const int* __restrict__ rowp, const int* __restrict__ eid,
    const float* __restrict__ bias, float* __restrict__ out, int E)
{
    __shared__ float sw[D];
    __shared__ int   ssrc[D];
    __shared__ float red[D];

    const int n   = blockIdx.x;
    const int tid = threadIdx.x;
    const int rb  = rowp[n];
    const int re  = rowp[n + 1];
    const float ad = adst[n];

    // pass 1: segment max
    float lm = -1e30f;
    for (int j = rb + tid; j < re; j += D) {
        int ei = eid[j];
        int s  = (ei < E) ? (int)srcArr[ei] : (ei - E);
        float e = asrc[s] + ad;
        e = (e > 0.f) ? e : NEG_SLOPE * e;
        lm = fmaxf(lm, e);
    }
    red[tid] = lm; __syncthreads();
    #pragma unroll
    for (int s = D / 2; s > 0; s >>= 1) {
        if (tid < s) red[tid] = fmaxf(red[tid], red[tid + s]);
        __syncthreads();
    }
    const float m = red[0];
    __syncthreads();

    // pass 2: segment sum of exp
    float ls = 0.f;
    for (int j = rb + tid; j < re; j += D) {
        int ei = eid[j];
        int s  = (ei < E) ? (int)srcArr[ei] : (ei - E);
        float e = asrc[s] + ad;
        e = (e > 0.f) ? e : NEG_SLOPE * e;
        ls += __expf(e - m);
    }
    red[tid] = ls; __syncthreads();
    #pragma unroll
    for (int s = D / 2; s > 0; s >>= 1) {
        if (tid < s) red[tid] += red[tid + s];
        __syncthreads();
    }
    const float inv = 1.0f / red[0];
    __syncthreads();

    // pass 3: out[n,f] = sum_e alpha_e * h[src_e, f]
    float acc = 0.f;
    for (int t0 = rb; t0 < re; t0 += D) {
        int j = t0 + tid;
        if (j < re) {
            int ei = eid[j];
            int s  = (ei < E) ? (int)srcArr[ei] : (ei - E);
            float e = asrc[s] + ad;
            e = (e > 0.f) ? e : NEG_SLOPE * e;
            sw[tid]   = __expf(e - m) * inv;
            ssrc[tid] = s;
        }
        __syncthreads();
        int cnt = min(D, re - t0);
        for (int t = 0; t < cnt; ++t)
            acc += sw[t] * h[(size_t)ssrc[t] * D + tid];
        __syncthreads();
    }
    float o = acc + bias[tid];
    out[(size_t)n * D + tid] = RELU ? fmaxf(o, 0.f) : o;
}

// ---------------------------------------------------------------------------
// Layer 3: output dim = 2 (too small for WMMA). Fused GEMM + alpha projection,
// one wave per node.
// ---------------------------------------------------------------------------
__global__ __launch_bounds__(256) void layer3_proj(
    const float* __restrict__ g2, const float* __restrict__ W3,
    const float* __restrict__ a3s, const float* __restrict__ a3d,
    float* __restrict__ h3, float* __restrict__ asrc, float* __restrict__ adst,
    int Nn, int K)
{
    const int lane = threadIdx.x & 31;
    const int node = (blockIdx.x * blockDim.x + threadIdx.x) >> 5;
    if (node >= Nn) return;
    float p0 = 0.f, p1 = 0.f;
    for (int k = lane; k < K; k += 32) {
        float v = g2[(size_t)node * K + k];
        p0 += v * W3[2 * k];
        p1 += v * W3[2 * k + 1];
    }
    p0 = waveRedSum(p0);
    p1 = waveRedSum(p1);
    if (lane == 0) {
        h3[2 * node]     = p0;
        h3[2 * node + 1] = p1;
        asrc[node] = p0 * a3s[0] + p1 * a3s[1];
        adst[node] = p0 * a3d[0] + p1 * a3d[1];
    }
}

// Layer-3 aggregation + bias + final row softmax (2 classes); one wave/node.
__global__ __launch_bounds__(256) void layer3_aggregate(
    const float* __restrict__ h3, const float* __restrict__ asrc,
    const float* __restrict__ adst, const long long* __restrict__ srcArr,
    const int* __restrict__ rowp, const int* __restrict__ eid,
    const float* __restrict__ bias, float* __restrict__ out, int E, int Nn)
{
    const int lane = threadIdx.x & 31;
    const int node = (blockIdx.x * blockDim.x + threadIdx.x) >> 5;
    if (node >= Nn) return;
    const int rb = rowp[node], re = rowp[node + 1];
    const float ad = adst[node];

    float lm = -1e30f;
    for (int j = rb + lane; j < re; j += 32) {
        int ei = eid[j];
        int s  = (ei < E) ? (int)srcArr[ei] : (ei - E);
        float e = asrc[s] + ad;
        e = (e > 0.f) ? e : NEG_SLOPE * e;
        lm = fmaxf(lm, e);
    }
    const float m = waveRedMax(lm);

    float ls = 0.f, a0 = 0.f, a1 = 0.f;
    for (int j = rb + lane; j < re; j += 32) {
        int ei = eid[j];
        int s  = (ei < E) ? (int)srcArr[ei] : (ei - E);
        float e = asrc[s] + ad;
        e = (e > 0.f) ? e : NEG_SLOPE * e;
        float w = __expf(e - m);
        ls += w;
        a0 += w * h3[2 * s];
        a1 += w * h3[2 * s + 1];
    }
    ls = waveRedSum(ls);
    a0 = waveRedSum(a0);
    a1 = waveRedSum(a1);
    if (lane == 0) {
        float inv = 1.0f / ls;
        float o0 = a0 * inv + bias[0];
        float o1 = a1 * inv + bias[1];
        float mm = fmaxf(o0, o1);
        float e0 = __expf(o0 - mm), e1 = __expf(o1 - mm);
        float d = e0 + e1;
        out[2 * node]     = e0 / d;
        out[2 * node + 1] = e1 / d;
    }
}

// ---------------------------------------------------------------------------
// Host-side launcher
// ---------------------------------------------------------------------------
extern "C" void kernel_launch(void* const* d_in, const int* in_sizes, int n_in,
                              void* d_out, int out_size, void* d_ws, size_t ws_size,
                              hipStream_t stream)
{
    (void)n_in; (void)out_size; (void)ws_size;
    // inputs: 0:x 1:edge_index 2:batch 3:W1 4:a1s 5:a1d 6:b1
    //         7:W2 8:a2s 9:a2d 10:b2 11:W3 12:a3s 13:a3d 14:b3
    const float*     x    = (const float*)d_in[0];
    const long long* eidx = (const long long*)d_in[1];
    const float* W1 = (const float*)d_in[3];
    const float* a1s = (const float*)d_in[4];
    const float* a1d = (const float*)d_in[5];
    const float* b1  = (const float*)d_in[6];
    const float* W2  = (const float*)d_in[7];
    const float* a2s = (const float*)d_in[8];
    const float* a2d = (const float*)d_in[9];
    const float* b2  = (const float*)d_in[10];
    const float* W3  = (const float*)d_in[11];
    const float* a3s = (const float*)d_in[12];
    const float* a3d = (const float*)d_in[13];
    const float* b3  = (const float*)d_in[14];

    const int d1  = in_sizes[4];               // 128
    const int fin = in_sizes[3] / d1;          // 784
    const int N   = in_sizes[0] / fin;         // 100000
    const int d2  = in_sizes[8];               // 256
    const int E   = in_sizes[1] / 2;           // 1600000
    const int ET  = E + N;                     // edges incl. self-loops
    const long long* src = eidx;
    const long long* dst = eidx + E;

    // workspace carve-out
    char* ws = (char*)d_ws;
    size_t off = 0;
    auto carve = [&](size_t bytes) -> void* {
        void* p = ws + off;
        off = (off + bytes + 255) & ~(size_t)255;
        return p;
    };
    float* h1     = (float*)carve((size_t)N * d1 * 4);
    float* g1     = (float*)carve((size_t)N * d1 * 4);
    float* h2     = (float*)carve((size_t)N * d2 * 4);
    float* g2     = (float*)carve((size_t)N * d2 * 4);
    float* h3     = (float*)carve((size_t)N * 2 * 4);
    float* asrc   = (float*)carve((size_t)N * 4);
    float* adst   = (float*)carve((size_t)N * 4);
    int*   deg    = (int*)carve((size_t)N * 4);
    int*   rowp   = (int*)carve((size_t)(N + 1) * 4);
    int*   cursor = (int*)carve((size_t)(N + 1) * 4);
    int*   eid    = (int*)carve((size_t)ET * 4);

    // ---- CSR by destination (built once, reused for all 3 layers) ----
    zero_int<<<(N + 255) / 256, 256, 0, stream>>>(deg, N);
    deg_count<<<(ET + 255) / 256, 256, 0, stream>>>(dst, deg, E, N);
    exscan<<<1, 1024, 0, stream>>>(deg, rowp, N);
    copy_int<<<(N + 255) / 256, 256, 0, stream>>>(rowp, cursor, N);
    fill_csr<<<(ET + 255) / 256, 256, 0, stream>>>(dst, cursor, eid, E, N);

    const int tiles_m32 = (N + 31) / 32;
    const int waveBlocks = (N + 7) / 8;   // 8 waves of 32 per 256-thread block

    // ---- layer 1: 784 -> 128 ----
    {
        int total = tiles_m32 * (d1 / 32);
        gemm_wmma_bf16<<<(total + 7) / 8, 256, 0, stream>>>(x, W1, h1, N, fin, d1);
        alpha_proj<<<waveBlocks, 256, 0, stream>>>(h1, a1s, a1d, asrc, adst, N, d1);
        if (d1 == 128)
            gat_aggregate<128, true><<<N, 128, 0, stream>>>(h1, asrc, adst, src, rowp, eid, b1, g1, E);
        else if (d1 == 256)
            gat_aggregate<256, true><<<N, 256, 0, stream>>>(h1, asrc, adst, src, rowp, eid, b1, g1, E);
    }

    // ---- layer 2: 128 -> 256 ----
    {
        int total = tiles_m32 * (d2 / 32);
        gemm_wmma_bf16<<<(total + 7) / 8, 256, 0, stream>>>(g1, W2, h2, N, d1, d2);
        alpha_proj<<<waveBlocks, 256, 0, stream>>>(h2, a2s, a2d, asrc, adst, N, d2);
        if (d2 == 256)
            gat_aggregate<256, true><<<N, 256, 0, stream>>>(h2, asrc, adst, src, rowp, eid, b2, g2, E);
        else if (d2 == 128)
            gat_aggregate<128, true><<<N, 128, 0, stream>>>(h2, asrc, adst, src, rowp, eid, b2, g2, E);
    }

    // ---- layer 3: 256 -> 2, then row softmax ----
    layer3_proj<<<waveBlocks, 256, 0, stream>>>(g2, W3, a3s, a3d, h3, asrc, adst, N, d2);
    layer3_aggregate<<<waveBlocks, 256, 0, stream>>>(h3, asrc, adst, src, rowp, eid, b3,
                                                     (float*)d_out, E, N);
}
